// CausalAttention_8409545966028
// MI455X (gfx1250) — matrix operand order repaired and checked by reading, not compiled, not run on hardware
//
#include <hip/hip_runtime.h>
#include <math.h>

#define D_MODEL 768
#define SEQ     2048
#define BATCH   4

typedef __bf16 bf16_t;
typedef __attribute__((ext_vector_type(8)))  float        v8f;
typedef __attribute__((ext_vector_type(16))) float        v16f;
typedef __attribute__((ext_vector_type(8)))  __bf16       v8bf;
typedef __attribute__((ext_vector_type(16))) __bf16       v16bf;
typedef __attribute__((ext_vector_type(4)))  unsigned int v4u;
typedef __attribute__((ext_vector_type(8)))  int          v8i;
typedef __attribute__((ext_vector_type(4)))  int          v4i;

static __device__ __forceinline__ v16bf cat16(v8bf lo, v8bf hi) {
    return __builtin_shufflevector(lo, hi, 0,1,2,3,4,5,6,7,8,9,10,11,12,13,14,15);
}

// ---------------------------------------------------------------------------
// Kernel 1: QKV projection (bf16 WMMA, f32 accumulate).
//   out[n][o] = sum_d x[n][d] * W[o][d]
// One wave computes one 16x16 output tile; K-loop over D=768 in steps of 32.
// Q,K stored row-major bf16; V stored transposed (Vt[b][d][seq]) so the PV
// GEMM's B-fragments are contiguous.
// ---------------------------------------------------------------------------
__global__ void __launch_bounds__(128)
qkv_proj_kernel(const float* __restrict__ x,
                const float* __restrict__ Wq,
                const float* __restrict__ Wk,
                const float* __restrict__ Wv,
                bf16_t* __restrict__ qb,
                bf16_t* __restrict__ kb,
                bf16_t* __restrict__ vtb)
{
    const int wave = threadIdx.x >> 5;
    const int lane = threadIdx.x & 31;
    const int m    = lane & 15;     // A row / B col / C col
    const int half = lane >> 4;

    const int row0  = blockIdx.x * 16;                 // over B*SEQ = 8192
    const int col0  = (blockIdx.y * 4 + wave) * 16;    // over D_OUT = 768
    const int which = blockIdx.z;                      // 0=Q 1=K 2=V

    const float* __restrict__ W = (which == 0) ? Wq : (which == 1) ? Wk : Wv;

    const float* xrow = x + (size_t)(row0 + m) * D_MODEL;
    const float* wrow = W + (size_t)(col0 + m) * D_MODEL;

    v8f acc = {0.f,0.f,0.f,0.f,0.f,0.f,0.f,0.f};
    for (int d = 0; d < D_MODEL; d += 32) {
        // A fragment: row m of x, K-chunk [half*16, +16) contiguous
        v8f f0 = *(const v8f*)(xrow + d + half * 16);
        v8f f1 = *(const v8f*)(xrow + d + half * 16 + 8);
        v16bf a = cat16(__builtin_convertvector(f0, v8bf),
                        __builtin_convertvector(f1, v8bf));
        // B fragment: B[k][n] = W[col0+n][d+k]; lane n holds 16 contiguous k
        v16f wf = *(const v16f*)(wrow + d + half * 16);
        v16bf b = __builtin_convertvector(wf, v16bf);
        acc = __builtin_amdgcn_wmma_f32_16x16x32_bf16(
            false, a, false, b, (short)0, acc, false, false);
    }

    if (which < 2) {
        bf16_t* __restrict__ dst = (which == 0) ? qb : kb;
#pragma unroll
        for (int r = 0; r < 8; ++r) {
            const int row = row0 + r + half * 8;
            dst[(size_t)row * D_MODEL + col0 + m] = (bf16_t)acc[r];
        }
    } else {
#pragma unroll
        for (int r = 0; r < 8; ++r) {
            const int row = row0 + r + half * 8;   // row in B*SEQ
            const int bb  = row >> 11;             // SEQ = 2048
            const int s   = row & (SEQ - 1);
            vtb[((size_t)bb * D_MODEL + col0 + m) * SEQ + s] = (bf16_t)acc[r];
        }
    }
}

// ---------------------------------------------------------------------------
// Kernel 2: causal attention with LDS-resident score rows.
// One workgroup (8 waves) per (batch, 16-query tile). 320KB LDS/WGP lets the
// full 16x2048 f32 score block + bf16 P block + Q tile stay resident.
// Q tile staged via the Tensor Data Mover (TENSOR_LOAD_TO_LDS).
// ---------------------------------------------------------------------------
#define LDS_Q_BYTES   (16 * D_MODEL * 2)            // 24576
#define LDS_S_BYTES   (16 * SEQ * 4)                // 131072
#define LDS_P_BYTES   (16 * SEQ * 2)                // 65536
#define LDS_S_OFF     LDS_Q_BYTES
#define LDS_P_OFF     (LDS_Q_BYTES + LDS_S_BYTES)
#define LDS_SUM_OFF   (LDS_Q_BYTES + LDS_S_BYTES + LDS_P_BYTES)
#define LDS_BYTES     (LDS_SUM_OFF + 64)

__global__ void __launch_bounds__(256)
causal_attn_kernel(const bf16_t* __restrict__ qb,
                   const bf16_t* __restrict__ kb,
                   const bf16_t* __restrict__ vtb,
                   float* __restrict__ out)
{
    extern __shared__ char smem[];
    bf16_t* sQ   = (bf16_t*)(smem);
    float*  sS   = (float* )(smem + LDS_S_OFF);
    bf16_t* sP   = (bf16_t*)(smem + LDS_P_OFF);
    float*  sSum = (float* )(smem + LDS_SUM_OFF);

    const int tid  = threadIdx.x;
    const int wave = tid >> 5;
    const int lane = tid & 31;
    const int m    = lane & 15;
    const int half = lane >> 4;

    const int qt = blockIdx.x & 127;      // SEQ/16 = 128 query tiles
    const int b  = blockIdx.x >> 7;
    const int q0 = qt * 16;

    // ---- stage Q tile (contiguous 16*768 bf16 = 24KB) via TDM ----
    // D# per CDNA5 ISA ch.8: 1D tensor, data_size=2B, one 12288-element row,
    // lds_addr=0 (start of this workgroup's LDS), type=2, count=1.
    if (wave == 0) {
        const unsigned long long ga =
            (unsigned long long)(qb + (size_t)(b * SEQ + q0) * D_MODEL);
        v4u g0;
        g0[0] = 1u;                                   // count=1, user mode
        g0[1] = 0u;                                   // lds_addr = 0 (sQ)
        g0[2] = (unsigned int)(ga & 0xffffffffu);     // global_addr[31:0]
        g0[3] = (unsigned int)((ga >> 32) & 0x01ffffffu) | (2u << 30); // type=2
        v8i g1;
        g1[0] = 0x00010000;          // workgroup_mask=0, data_size=1 (2 bytes)
        g1[1] = 0x3000 << 16;        // tensor_dim0[15:0]=12288 in bits[63:48]
        g1[2] = 1 << 16;             // tensor_dim0 hi=0; tensor_dim1[15:0]=1
        g1[3] = 0x3000 << 16;        // tensor_dim1 hi=0; tile_dim0=12288
        g1[4] = 1;                   // tile_dim1=1, tile_dim2=0
        g1[5] = 0x3000;              // tensor_dim0_stride = 12288
        g1[6] = 0;                   // stride hi, tensor_dim1_stride lo
        g1[7] = 0;
        v4i gz4 = {0, 0, 0, 0};      // groups 2/3 unused (<=2D tensor)
        v8i gz8 = {0, 0, 0, 0, 0, 0, 0, 0};
        __builtin_amdgcn_tensor_load_to_lds(g0, g1, gz4, gz4, gz8, 0);
        __builtin_amdgcn_s_wait_tensorcnt(0);
    }
    __syncthreads();

    // ---- Phase 1: S = (Q K^T) * scale, causal masked, into LDS ----
    const float scale = 0.036084391824351615f;   // 1/sqrt(768)
    const int nkt = qt + 1;                       // key tiles 0..qt (causal)
    for (int kt = wave; kt < nkt; kt += 8) {
        v8f s = {0.f,0.f,0.f,0.f,0.f,0.f,0.f,0.f};
        const bf16_t* qrow = sQ + m * D_MODEL;
        const bf16_t* krow = kb + (size_t)(b * SEQ + kt * 16 + m) * D_MODEL;
        if (kt + 8 < nkt)            // prefetch this lane's row in next tile
            __builtin_prefetch(krow + (size_t)8 * 16 * D_MODEL, 0, 1);
        for (int d = 0; d < D_MODEL; d += 32) {
            v16bf a  = cat16(*(const v8bf*)(qrow + d + half * 16),
                             *(const v8bf*)(qrow + d + half * 16 + 8));
            v16bf bb = *(const v16bf*)(krow + d + half * 16);
            s = __builtin_amdgcn_wmma_f32_16x16x32_bf16(
                false, a, false, bb, (short)0, s, false, false);
        }
        const int key = kt * 16 + m;
#pragma unroll
        for (int r = 0; r < 8; ++r) {
            const int qi = r + half * 8;
            float v = s[r] * scale;
            if (key > q0 + qi) v = -__builtin_inff();
            sS[qi * SEQ + key] = v;
        }
    }
    __syncthreads();

    // ---- Phase 2: row softmax (16 threads per row, shfl_xor reductions) ----
    {
        const int row = tid >> 4;       // 0..15
        const int t   = tid & 15;
        const int L   = q0 + 16;        // valid key count
        const int Lp  = (L + 31) & ~31; // padded to WMMA K granularity
        float mx = -__builtin_inff();
        for (int j = t; j < L; j += 16) mx = fmaxf(mx, sS[row * SEQ + j]);
        for (int off = 8; off; off >>= 1) mx = fmaxf(mx, __shfl_xor(mx, off, 32));
        float sum = 0.f;
        for (int j = t; j < Lp; j += 16) {
            const float e = (j < L) ? __expf(sS[row * SEQ + j] - mx) : 0.f;
            sum += e;
            sP[row * SEQ + j] = (bf16_t)e;   // unnormalized P
        }
        for (int off = 8; off; off >>= 1) sum += __shfl_xor(sum, off, 32);
        if (t == 0) sSum[row] = sum;
    }
    __syncthreads();

    // ---- Phase 3: O = (P V) / rowsum; each wave owns 96 output dims ----
    const int L  = q0 + 16;
    const int Lp = (L + 31) & ~31;
#pragma unroll
    for (int tt = 0; tt < 6; ++tt) {
        const int n0 = wave * 96 + tt * 16;
        v8f acc = {0.f,0.f,0.f,0.f,0.f,0.f,0.f,0.f};
        const bf16_t* prow = sP + m * SEQ;
        const bf16_t* vrow = vtb + ((size_t)b * D_MODEL + n0 + m) * SEQ;
        for (int kc = 0; kc < Lp; kc += 32) {
            if (kc + 512 < Lp) __builtin_prefetch(vrow + kc + 512, 0, 1);
            v16bf a  = cat16(*(const v8bf*)(prow + kc + half * 16),
                             *(const v8bf*)(prow + kc + half * 16 + 8));
            v16bf bb = *(const v16bf*)(vrow + kc + half * 16);
            acc = __builtin_amdgcn_wmma_f32_16x16x32_bf16(
                false, a, false, bb, (short)0, acc, false, false);
        }
#pragma unroll
        for (int r = 0; r < 8; ++r) {
            const int qi = r + half * 8;
            out[(size_t)(b * SEQ + q0 + qi) * D_MODEL + n0 + m] = acc[r] / sSum[qi];
        }
    }
}

// ---------------------------------------------------------------------------
extern "C" void kernel_launch(void* const* d_in, const int* in_sizes, int n_in,
                              void* d_out, int out_size, void* d_ws, size_t ws_size,
                              hipStream_t stream)
{
    (void)in_sizes; (void)n_in; (void)out_size; (void)ws_size;
    const float* x  = (const float*)d_in[0];
    const float* Wq = (const float*)d_in[1];
    const float* Wk = (const float*)d_in[2];
    const float* Wv = (const float*)d_in[3];
    float* out = (float*)d_out;

    const size_t elems = (size_t)BATCH * SEQ * D_MODEL;
    bf16_t* qb  = (bf16_t*)d_ws;
    bf16_t* kb  = qb + elems;
    bf16_t* vtb = kb + elems;     // stored as [B][D][SEQ]

    // QKV projection: (8192/16) row tiles x (768/64) col-tile groups x {Q,K,V}
    dim3 g1((BATCH * SEQ) / 16, D_MODEL / 64, 3);
    qkv_proj_kernel<<<g1, 128, 0, stream>>>(x, Wq, Wk, Wv, qb, kb, vtb);

    // Attention: one WG per (batch, 16-query tile), 216KB dynamic LDS.
    (void)hipFuncSetAttribute((const void*)causal_attn_kernel,
                              hipFuncAttributeMaxDynamicSharedMemorySize,
                              LDS_BYTES);
    causal_attn_kernel<<<dim3(BATCH * (SEQ / 16)), 256, LDS_BYTES, stream>>>(
        qb, kb, vtb, out);
}
